// Head_51204600103251
// MI455X (gfx1250) — matrix operand order repaired and checked by reading, not compiled, not run on hardware
//
#include <hip/hip_runtime.h>
#include <hip/hip_bf16.h>

// ---- types for WMMA fragments (wave32, CDNA5) ----
typedef __attribute__((ext_vector_type(16))) __bf16          v16bf;
typedef __attribute__((ext_vector_type(16))) unsigned short  v16u;
typedef __attribute__((ext_vector_type(8)))  float           v8f;

#define DFL_CH  16
#define NC      20
#define COUT    84          // 64 box + 20 cls
#define CPAD    96          // padded to 6 WMMA column tiles
#define KDIM    128
#define NTILE   6           // 96 / 16
#define KSTEP   4           // 128 / 32
#define WAVES   4           // waves per block (wave32) -> 128 threads
#define HW      6400
#define WIMG    80

#define NFRAG   (NTILE * KSTEP)            // 24 B fragments
#define BWORDS  (NFRAG * 32 * 16)          // 12288 ushorts = 24 KB
#define BVEC4   (BWORDS / 8)               // 1536 uint4

__device__ __forceinline__ unsigned short f2bf(float f) {
    // round-to-nearest-even f32 -> bf16 (used by pack kernel only)
    unsigned u = __float_as_uint(f);
    u += 0x7FFFu + ((u >> 16) & 1u);
    return (unsigned short)(u >> 16);
}

// ---------------------------------------------------------------------------
// Pack W = [w_box | w_cls | 0-pad] (128 x 96) into bf16 WMMA B-fragments.
// Fragment (n, kk): B tile = K rows [kk*32, kk*32+32) x N cols [n*16, n*16+16).
// B-matrix 32x16 bf16 layout: lane&15 = N, element j -> K = 8*(lane>>4) + j (+8 if j>=8).
// Also packs bias (96 f32, zero-padded).
// ---------------------------------------------------------------------------
__global__ void head_pack_kernel(const float* __restrict__ w_box,
                                 const float* __restrict__ b_box,
                                 const float* __restrict__ w_cls,
                                 const float* __restrict__ b_cls,
                                 unsigned short* __restrict__ wsB,
                                 float* __restrict__ wsBias) {
    const int tid  = threadIdx.x;          // 0..767
    const int fi   = tid >> 5;             // fragment id = n*4 + kk  (0..23)
    const int lane = tid & 31;
    const int n    = fi >> 2;
    const int kk   = fi & 3;
    const int k0   = kk * 32 + (lane >> 4) * 8;   // B: lane&15 indexes N, element indexes K

    unsigned short* dst = wsB + ((size_t)fi * 32 + lane) * 16;
#pragma unroll
    for (int j = 0; j < 16; ++j) {
        const int col = n * 16 + (lane & 15);
        const int k   = k0 + j + (j >= 8 ? 8 : 0);
        float v = 0.0f;
        if (col < 64)            v = w_box[(size_t)k * 64 + col];
        else if (col < COUT)     v = w_cls[(size_t)k * NC + (col - 64)];
        dst[j] = f2bf(v);
    }

    if (tid < CPAD) {
        float bv = 0.0f;
        if (tid < 64)           bv = b_box[tid];
        else if (tid < COUT)    bv = b_cls[tid - 64];
        wsBias[tid] = bv;
    }
}

// ---------------------------------------------------------------------------
// Main head kernel: 4 waves / block, 16 rows / wave.
//   - stage all 24 bf16 B-fragments in LDS once per block
//   - load all 64 x-floats per lane in one clause, convert to 4 A-fragments
//   - 24 x v_wmma_f32_16x16x32_bf16 with only LDS reads in the loop
//   - epilogue: DFL softmax-EV + anchor decode + sigmoid(cls) via LDS
// ---------------------------------------------------------------------------
__global__ void __launch_bounds__(128)
head_main_kernel(const float* __restrict__ x,
                 const unsigned short* __restrict__ wsB,
                 const float* __restrict__ wsBias,
                 const int* __restrict__ stride_p,
                 float* __restrict__ out) {
    __shared__ uint4  ldsB4[BVEC4];            // 24 KB packed B fragments
    __shared__ float  tileC[WAVES][16][CPAD];  // 24 KB
    __shared__ float  fbuf[WAVES][16][4];      // 1 KB

    const unsigned short* ldsB = (const unsigned short*)ldsB4;

    const int lane  = threadIdx.x & 31;
    const int wv    = threadIdx.x >> 5;
    const int tile  = blockIdx.x * WAVES + wv; // 16-row tile index
    const int r0    = tile * 16;
    const int row16 = lane & 15;               // A-matrix: row within tile
    const int hi    = lane >> 4;               // 0: K 0-7/16-23, 1: K 8-15/24-31

    // ---- stage B fragments into LDS (all 128 threads cooperate) ----
    {
        const uint4* src = (const uint4*)wsB;
#pragma unroll
        for (int i = 0; i < BVEC4 / 128; ++i)  // 12 iterations
            ldsB4[threadIdx.x + i * 128] = src[threadIdx.x + i * 128];
    }

    // ---- load this lane's 64 x-floats (half-row of 2 lanes) in one clause ----
    const size_t xbase = (size_t)(r0 + row16) * KDIM + (size_t)hi * 8;
    float4 xa[16];
#pragma unroll
    for (int kk = 0; kk < KSTEP; ++kk) {
        const float4* p0 = (const float4*)(x + xbase + (size_t)kk * 32);
        const float4* p1 = (const float4*)(x + xbase + (size_t)kk * 32 + 16);
        xa[kk * 4 + 0] = p0[0];
        xa[kk * 4 + 1] = p0[1];
        xa[kk * 4 + 2] = p1[0];
        xa[kk * 4 + 3] = p1[1];
    }

    // ---- convert to 4 bf16 A-fragments (native cvt) ----
    // A 16x32 bf16 layout: elems 0..7 -> K=kbase+0..7, elems 8..15 -> K=kbase+16..23
    v16bf af[KSTEP];
#define CVT4(V, B, F4)                                                         \
    V[B + 0] = (__bf16)(F4).x; V[B + 1] = (__bf16)(F4).y;                      \
    V[B + 2] = (__bf16)(F4).z; V[B + 3] = (__bf16)(F4).w;
#pragma unroll
    for (int kk = 0; kk < KSTEP; ++kk) {
        CVT4(af[kk], 0,  xa[kk * 4 + 0]);
        CVT4(af[kk], 4,  xa[kk * 4 + 1]);
        CVT4(af[kk], 8,  xa[kk * 4 + 2]);
        CVT4(af[kk], 12, xa[kk * 4 + 3]);
    }
#undef CVT4

    __syncthreads();   // B fragments staged

    // ---- GEMM: 24 WMMAs, 6 independent accumulator chains ----
    v8f acc[NTILE];
#pragma unroll
    for (int n = 0; n < NTILE; ++n) {
        v8f z = {0.f, 0.f, 0.f, 0.f, 0.f, 0.f, 0.f, 0.f};
        acc[n] = z;
    }

#pragma unroll
    for (int kk = 0; kk < KSTEP; ++kk) {
#pragma unroll
        for (int n = 0; n < NTILE; ++n) {
            const v16u bu =
                *(const v16u*)(ldsB + ((n * 4 + kk) * 32 + lane) * 16);
            acc[n] = __builtin_amdgcn_wmma_f32_16x16x32_bf16(
                false, af[kk], false, __builtin_bit_cast(v16bf, bu),
                (short)0, acc[n], false, false);
        }
    }

    // ---- spill C + bias to LDS ----
    // C/D layout: VGPR v, lane l -> row = v + (l>=16 ? 8 : 0), col = n*16 + (l&15)
#pragma unroll
    for (int n = 0; n < NTILE; ++n) {
        const int col = n * 16 + (lane & 15);
        const float bi = wsBias[col];
#pragma unroll
        for (int v = 0; v < 8; ++v) {
            tileC[wv][v + hi * 8][col] = acc[n][v] + bi;
        }
    }
    __syncthreads();

    // ---- DFL: softmax expected value over 4 groups of 16 bins ----
    // lane handles row = lane&15, fields {hi*2, hi*2+1}
    {
        const int r = lane & 15;
#pragma unroll
        for (int t = 0; t < 2; ++t) {
            const int f = hi * 2 + t;
            const float* p = &tileC[wv][r][f * 16];
            float m = p[0];
#pragma unroll
            for (int i = 1; i < 16; ++i) m = fmaxf(m, p[i]);
            float s = 0.f, ev = 0.f;
#pragma unroll
            for (int i = 0; i < 16; ++i) {
                const float e = __expf(p[i] - m);
                s  += e;
                ev += e * (float)i;
            }
            fbuf[wv][r][f] = ev * __builtin_amdgcn_rcpf(s);
        }
    }
    __syncthreads();

    const float sflt = (float)stride_p[0];

    // ---- anchor decode: lanes 0..15, one row each ----
    if (lane < 16) {
        const int ng  = r0 + lane;
        const int pix = ng % HW;
        const float ax = ((float)(pix % WIMG) + 0.5f) * sflt;
        const float ay = ((float)(pix / WIMG) + 0.5f) * sflt;
        const float f0 = fbuf[wv][lane][0];
        const float f1 = fbuf[wv][lane][1];
        const float f2 = fbuf[wv][lane][2];
        const float f3 = fbuf[wv][lane][3];
        float4 o;
        o.x = ax + 0.5f * (f2 - f0);
        o.y = ay + 0.5f * (f3 - f1);
        o.z = f2 - f0;
        o.w = f3 - f1;
        *(float4*)(out + (size_t)ng * 24) = o;   // 24 floats/row -> 16B aligned
    }

    // ---- sigmoid on 20 class logits: 16 rows x 20 cols = 320 = 32 lanes x 10 ----
#pragma unroll
    for (int j = 0; j < 10; ++j) {
        const int idx = lane * 10 + j;
        const int r   = idx / NC;
        const int c   = idx % NC;
        const float v = tileC[wv][r][64 + c];
        out[(size_t)(r0 + r) * 24 + 4 + c] =
            __builtin_amdgcn_rcpf(1.0f + __expf(-v));
    }
}

extern "C" void kernel_launch(void* const* d_in, const int* in_sizes, int n_in,
                              void* d_out, int out_size, void* d_ws, size_t ws_size,
                              hipStream_t stream) {
    const float* x     = (const float*)d_in[0];
    const float* w_box = (const float*)d_in[1];
    const float* b_box = (const float*)d_in[2];
    const float* w_cls = (const float*)d_in[3];
    const float* b_cls = (const float*)d_in[4];
    const int*   strd  = (const int*)d_in[5];

    // workspace: 24 fragments * 32 lanes * 16 bf16 = 24 KB, then 96 f32 bias
    unsigned short* wsB    = (unsigned short*)d_ws;
    float*          wsBias = (float*)((char*)d_ws + (size_t)BWORDS * sizeof(unsigned short));

    head_pack_kernel<<<1, 768, 0, stream>>>(w_box, b_box, w_cls, b_cls, wsB, wsBias);

    // 204800 rows = 3200 blocks * 4 waves * 16 rows (exact)
    head_main_kernel<<<3200, 128, 0, stream>>>(x, wsB, wsBias, strd, (float*)d_out);
}
